// kopt_Decoder_14637248544936
// MI455X (gfx1250) — matrix operand order repaired and checked by reading, not compiled, not run on hardware
//
#include <hip/hip_runtime.h>
#include <math.h>

#define BSZ 128
#define GS  2000
#define EE  128
#define KMAX 4
#define NEGV (-1e30f)
#define CLIPV 10.0f

// LDS weight tile: 4 matrices * 128 rows, each row 64 dwords of bf16 pairs,
// padded to 68 dwords so the 16 f-lanes of a B-fragment read hit distinct banks.
#define WPITCH 68
#define WROWS  (4 * EE)
#define WLDS_DW (WROWS * WPITCH)

typedef __attribute__((ext_vector_type(16))) __bf16 v16bf;
typedef __attribute__((ext_vector_type(8)))  float  v8f;
typedef __attribute__((ext_vector_type(8)))  int    v8i;

__device__ __forceinline__ unsigned int f2bf(float x) {
  unsigned int u = __float_as_uint(x);
  return (u + 0x7FFFu + ((u >> 16) & 1u)) >> 16;   // round-to-nearest-even bf16
}

__device__ __forceinline__ v8f wmma_bf16(v16bf a, v16bf b, v8f c) {
  return __builtin_amdgcn_wmma_f32_16x16x32_bf16(false, a, false, b, (short)0, c, false, false);
}

// Batched hardware tanh (CDNA5 V_TANH_F32). Eight independent trans ops cover
// each other's TRANS-result hazard; single trailing v_nop covers the last one.
__device__ __forceinline__ void tanh8(float* x) {
  asm volatile(
      "v_tanh_f32 %0, %0\n\t"
      "v_tanh_f32 %1, %1\n\t"
      "v_tanh_f32 %2, %2\n\t"
      "v_tanh_f32 %3, %3\n\t"
      "v_tanh_f32 %4, %4\n\t"
      "v_tanh_f32 %5, %5\n\t"
      "v_tanh_f32 %6, %6\n\t"
      "v_tanh_f32 %7, %7\n\t"
      "v_nop"
      : "+v"(x[0]), "+v"(x[1]), "+v"(x[2]), "+v"(x[3]),
        "+v"(x[4]), "+v"(x[5]), "+v"(x[6]), "+v"(x[7]));
}

__device__ __forceinline__ float sigmoidf_(float x) { return 1.0f / (1.0f + __expf(-x)); }

// ---------------------------------------------------------------- prep: W_K* -> bf16
__global__ void k_prep(const float* __restrict__ w0, const float* __restrict__ w1,
                       const float* __restrict__ w2, const float* __restrict__ w3,
                       unsigned short* __restrict__ wK) {
  int idx = blockIdx.x * blockDim.x + threadIdx.x;
  if (idx >= 4 * EE * EE) return;
  const float* ws4[4] = {w0, w1, w2, w3};
  int mat = idx >> 14, r = idx & 16383;
  wK[idx] = (unsigned short)f2bf(ws4[mat][r]);
}

// ---------------------------------------------------------------- state init
__global__ void k_init(int* mask, int* stopped, int* nol, float* ll,
                       int* ai, int* kl, int* kr) {
  int idx = blockIdx.x * blockDim.x + threadIdx.x;
  if (idx < BSZ * GS) mask[idx] = 0;
  if (idx < BSZ) { stopped[idx] = 1; nol[idx] = -1; ll[idx] = 0.0f; }
  if (idx < BSZ * KMAX) { ai[idx] = 0; kr[idx] = 0; }
  if (idx < BSZ * (KMAX + 1)) kl[idx] = 0;
}

// ---------------------------------------------------------------- mean over g
__global__ __launch_bounds__(EE) void k_mean(const float* __restrict__ h, float* __restrict__ hmean) {
  int b = blockIdx.x, e = threadIdx.x;
  const float* hb = h + (size_t)b * GS * EE;
  float s = 0.0f;
  for (int g = 0; g < GS; ++g) s += hb[(size_t)g * EE + e];
  hmean[b * EE + e] = s * (1.0f / (float)GS);
}

// ---------------------------------------------------------------- q init
__global__ __launch_bounds__(EE) void k_qinit(const float* __restrict__ hmean,
                                              const float* __restrict__ W_init,
                                              const float* __restrict__ b_init,
                                              const float* __restrict__ init_query,
                                              float* q1, float* q2, float* iq1, float* iq2) {
  int b = blockIdx.x, f = threadIdx.x;
  __shared__ float hm[EE];
  hm[f] = hmean[b * EE + f];
  __syncthreads();
  float s = b_init[f];
  for (int e = 0; e < EE; ++e) s += hm[e] * W_init[f * EE + e];
  q1[b * EE + f] = s;
  q2[b * EE + f] = s;
  iq1[b * EE + f] = init_query[f];
  iq2[b * EE + f] = init_query[f];
}

// ---------------------------------------------------------------- GRU cell helper
__device__ __forceinline__ float gru_cell(const float* xs, const float* hs,
                                          const float* __restrict__ Wih, const float* __restrict__ Whh,
                                          const float* __restrict__ bih, const float* __restrict__ bhh,
                                          int f) {
  float gir = bih[f], giz = bih[EE + f], gin = bih[2 * EE + f];
  float ghr = bhh[f], ghz = bhh[EE + f], ghn = bhh[2 * EE + f];
  const float* wi0 = Wih + (size_t)f * EE;
  const float* wi1 = Wih + (size_t)(EE + f) * EE;
  const float* wi2 = Wih + (size_t)(2 * EE + f) * EE;
  const float* wh0 = Whh + (size_t)f * EE;
  const float* wh1 = Whh + (size_t)(EE + f) * EE;
  const float* wh2 = Whh + (size_t)(2 * EE + f) * EE;
  for (int e = 0; e < EE; ++e) {
    float xe = xs[e], he = hs[e];
    gir += wi0[e] * xe; giz += wi1[e] * xe; gin += wi2[e] * xe;
    ghr += wh0[e] * he; ghz += wh1[e] * he; ghn += wh2[e] * he;
  }
  float r = sigmoidf_(gir + ghr);
  float z = sigmoidf_(giz + ghz);
  float n = tanhf(gin + r * ghn);
  return (1.0f - z) * n + z * hs[f];
}

// ---------------------------------------------------------------- GRU step + query projections
__global__ __launch_bounds__(EE) void k_gru(
    const float* __restrict__ Wih1, const float* __restrict__ Whh1,
    const float* __restrict__ bih1, const float* __restrict__ bhh1,
    const float* __restrict__ Wih2, const float* __restrict__ Whh2,
    const float* __restrict__ bih2, const float* __restrict__ bhh2,
    const float* __restrict__ WQ1, const float* __restrict__ WQ2,
    const float* __restrict__ WQ3, const float* __restrict__ WQ4,
    const float* __restrict__ iq1, const float* __restrict__ iq2,
    float* q1, float* q2, float* A1, float* B1, float* A2, float* B2) {
  int b = blockIdx.x, f = threadIdx.x;
  __shared__ float xs[EE], hs[EE], hn[EE];

  // rnn1
  xs[f] = iq1[b * EE + f];
  hs[f] = q1[b * EE + f];
  __syncthreads();
  float h1 = gru_cell(xs, hs, Wih1, Whh1, bih1, bhh1, f);
  __syncthreads();
  hn[f] = h1;
  q1[b * EE + f] = h1;
  __syncthreads();
  {
    float s1 = 0.0f, s3 = 0.0f;
    const float* r1 = WQ1 + (size_t)f * EE;
    const float* r3 = WQ3 + (size_t)f * EE;
    for (int e = 0; e < EE; ++e) { s1 += r1[e] * hn[e]; s3 += r3[e] * hn[e]; }
    A1[b * EE + f] = s1; B1[b * EE + f] = s3;
  }
  __syncthreads();

  // rnn2
  xs[f] = iq2[b * EE + f];
  hs[f] = q2[b * EE + f];
  __syncthreads();
  float h2 = gru_cell(xs, hs, Wih2, Whh2, bih2, bhh2, f);
  __syncthreads();
  hn[f] = h2;
  q2[b * EE + f] = h2;
  __syncthreads();
  {
    float s2 = 0.0f, s4 = 0.0f;
    const float* r2 = WQ2 + (size_t)f * EE;
    const float* r4 = WQ4 + (size_t)f * EE;
    for (int e = 0; e < EE; ++e) { s2 += r2[e] * hn[e]; s4 += r4[e] * hn[e]; }
    A2[b * EE + f] = s2; B2[b * EE + f] = s4;
  }
}

// ---------------------------------------------------------------- fused WMMA score kernel
// Weights (4x128x128 bf16) staged global->LDS with GLOBAL_LOAD_ASYNC_TO_LDS_B128;
// each wave owns a 16-row g-tile: A = bf16 h tile, B = bf16 W fragments preloaded
// from LDS (clause of ds_load_b128) feeding 8 back-to-back WMMAs on two
// independent accumulator chains; batched hardware v_tanh_f32 nonlinearity.
__global__ __launch_bounds__(256) void k_score(
    const float* __restrict__ h, const unsigned short* __restrict__ wK,
    const float* __restrict__ A1q, const float* __restrict__ B1q,
    const float* __restrict__ A2q, const float* __restrict__ B2q,
    const float* __restrict__ V1, const float* __restrict__ V2,
    float* __restrict__ logits) {
  extern __shared__ int lds_w[];            // WLDS_DW dwords (139264 bytes)
  const int b = blockIdx.x;
  const int tid = threadIdx.x;

  // ---- async stage: 512 rows x 64 dwords (16 b128 chunks/row) -> padded LDS rows
  {
    const int* wsrc = (const int*)wK;
#pragma unroll
    for (int k = 0; k < 32; ++k) {
      int chunk = tid + (k << 8);           // 8192 chunks total
      int row = chunk >> 4;
      int cw = (chunk & 15) << 2;           // dword offset in row
      unsigned lds_byte = (unsigned)((row * WPITCH + cw) << 2);
      const int* gp = wsrc + row * 64 + cw;
      asm volatile("global_load_async_to_lds_b128 %0, %1, off"
                   :: "v"(lds_byte), "v"(gp)
                   : "memory");
    }
    asm volatile("s_wait_asynccnt 0" ::: "memory");
    __syncthreads();
  }

  const int wave = tid >> 5;
  const int lane = tid & 31;
  const int tile = blockIdx.y * 8 + wave;
  if (tile >= GS / 16) return;              // uniform per wave: EXEC all-ones for WMMA
  const int g0 = tile * 16;
  const int m = lane & 15;
  const int half = lane >> 4;
  const float* hrow = h + ((size_t)b * GS + g0 + m) * EE;
  if (tile + 1 < GS / 16) __builtin_prefetch(hrow + 16 * EE, 0, 0);  // global_prefetch_b8

  // A fragments: 16x32 bf16 per k-tile, k = 16*half + 2*v
  v16bf afr[4];
#pragma unroll
  for (int kt = 0; kt < 4; ++kt) {
    v8i pk;
#pragma unroll
    for (int v = 0; v < 8; ++v) {
      int e = kt * 32 + (half << 4) + (v << 1);
      unsigned int lo = f2bf(hrow[e]);
      unsigned int hi = f2bf(hrow[e + 1]);
      pk[v] = (int)(lo | (hi << 16));
    }
    afr[kt] = __builtin_bit_cast(v16bf, pk);
  }

  float acc[8];
#pragma unroll
  for (int v = 0; v < 8; ++v) acc[v] = 0.0f;

  const float* A1b = A1q + b * EE;
  const float* B1b = B1q + b * EE;
  const float* A2b = A2q + b * EE;
  const float* B2b = B2q + b * EE;

#pragma unroll 1
  for (int nt = 0; nt < 8; ++nt) {
    const int f = nt * 16 + m;
    const float a1 = A1b[f], b1 = B1b[f], a2 = A2b[f], b2 = B2b[f];
    const float v1 = V1[f], v2 = V2[f];

    // two Kh tiles (matX, matY): preload all 8 B fragments (ds_load clause),
    // then 8 WMMAs back-to-back on two independent accumulator chains.
    auto ktile2 = [&](int matX, int matY, v8f& cX, v8f& cY) {
      const int rowX = (matX * EE + f) * WPITCH;
      const int rowY = (matY * EE + f) * WPITCH;
      v16bf bx[4], by[4];
#pragma unroll
      for (int kt = 0; kt < 4; ++kt) {
        const int off = kt * 16 + (half << 3);   // dword offset of 32B fragment slice
        const int4* pX = (const int4*)&lds_w[rowX + off];
        const int4* pY = (const int4*)&lds_w[rowY + off];
        int4 x0 = pX[0], x1 = pX[1];
        int4 y0 = pY[0], y1 = pY[1];
        v8i pkX = {x0.x, x0.y, x0.z, x0.w, x1.x, x1.y, x1.z, x1.w};
        v8i pkY = {y0.x, y0.y, y0.z, y0.w, y1.x, y1.y, y1.z, y1.w};
        bx[kt] = __builtin_bit_cast(v16bf, pkX);
        by[kt] = __builtin_bit_cast(v16bf, pkY);
      }
#pragma unroll
      for (int kt = 0; kt < 4; ++kt) {
        cX = wmma_bf16(afr[kt], bx[kt], cX);
        cY = wmma_bf16(afr[kt], by[kt], cY);
      }
    };

    float t[8];

    v8f d1 = {0.f, 0.f, 0.f, 0.f, 0.f, 0.f, 0.f, 0.f};
    v8f d3 = d1;
    ktile2(0, 2, d1, d3);
#pragma unroll
    for (int v = 0; v < 8; ++v) t[v] = d1[v] + a1 + d3[v] * b1;
    tanh8(t);
#pragma unroll
    for (int v = 0; v < 8; ++v) acc[v] += v1 * t[v];

    v8f d2 = {0.f, 0.f, 0.f, 0.f, 0.f, 0.f, 0.f, 0.f};
    v8f d4 = d2;
    ktile2(1, 3, d2, d4);
#pragma unroll
    for (int v = 0; v < 8; ++v) t[v] = d2[v] + a2 + d4[v] * b2;
    tanh8(t);
#pragma unroll
    for (int v = 0; v < 8; ++v) acc[v] += v2 * t[v];
  }

  // reduce over f (16 lanes per half); D layout: lane=col f, VGPR v = row m (+8 upper half)
#pragma unroll
  for (int off = 1; off < 16; off <<= 1) {
#pragma unroll
    for (int v = 0; v < 8; ++v) acc[v] += __shfl_xor(acc[v], off, 32);
  }
  if (m == 0) {
    tanh8(acc);
    float* Lg = logits + (size_t)b * GS + g0 + half * 8;
#pragma unroll
    for (int v = 0; v < 8; ++v) Lg[v] = acc[v] * CLIPV;
  }
}

// ---------------------------------------------------------------- per-batch select / state update
__global__ __launch_bounds__(256) void k_select(
    int iter, const float* __restrict__ h, const int* __restrict__ rec,
    const int* __restrict__ visited_time, const int* __restrict__ last_action,
    const int* __restrict__ fixed_action, const float* __restrict__ logits,
    int* mask, int* vtt, int* stopped_g, int* nol_g, int* ai, int* kl, int* kr,
    float* ll, float* iq1, float* iq2) {
  const int b = blockIdx.x, t = threadIdx.x;
  __shared__ float red[256];
  __shared__ float s_max, s_sum;
  __shared__ int s_action, s_stop_old, s_stop_new, s_nol_old, s_vtt_a, s_ai0, s_nna;

  const float* lg = logits + (size_t)b * GS;
  int* mk = mask + (size_t)b * GS;
  int* vt = vtt + (size_t)b * GS;
  const int la = last_action[b];

  auto ml = [&](int g) -> float {
    float x = lg[g];
    if (mk[g]) x = NEGV;
    if (iter == 0 && g == la) x = NEGV;
    return x;
  };

  // 1) max
  float mx = -3.0e38f;
  for (int g = t; g < GS; g += 256) mx = fmaxf(mx, ml(g));
  red[t] = mx;
  __syncthreads();
  for (int s = 128; s > 0; s >>= 1) {
    if (t < s) red[t] = fmaxf(red[t], red[t + s]);
    __syncthreads();
  }
  if (t == 0) s_max = red[0];
  __syncthreads();

  // 2) sum exp
  float sm = 0.0f;
  for (int g = t; g < GS; g += 256) sm += __expf(ml(g) - s_max);
  red[t] = sm;
  __syncthreads();
  for (int s = 128; s > 0; s >>= 1) {
    if (t < s) red[t] += red[t + s];
    __syncthreads();
  }
  if (t == 0) s_sum = red[0];
  __syncthreads();

  // 3) scalar state machine (thread 0)
  if (t == 0) {
    int stop = stopped_g[b];
    int nol = nol_g[b];
    int action = fixed_action[b * KMAX + iter];
    if (iter > 0 && stop) action = ai[b * KMAX + 0];
    float loss = ml(action) - s_max - __logf(s_sum);
    if (iter == 0) ll[b] = loss;
    else if (!stop) ll[b] += loss;
    int nna = rec[(size_t)b * GS + action];
    ai[b * KMAX + iter] = action;
    if (stop) kl[b * (KMAX + 1) + iter] = action;
    if (!stop) kr[b * KMAX + ((iter + 3) & 3)] = action;
    kl[b * (KMAX + 1) + iter + 1] = nna;
    int eq = (action == nol);
    int newstop = (iter == 0) ? eq : (stop | eq);
    if (newstop) kl[b * (KMAX + 1) + iter] = kl[b * (KMAX + 1) + ((iter + 4) % 5)];
    if (newstop) kr[b * KMAX + iter] = kr[b * KMAX + ((iter + 3) & 3)];
    s_action = action; s_stop_old = stop; s_stop_new = newstop;
    s_nol_old = nol; s_ai0 = ai[b * KMAX + 0]; s_nna = nna;
    stopped_g[b] = newstop;
    nol_g[b] = newstop ? -1 : nna;
  }
  __syncthreads();

  // 4) vtt (iter 0 only)
  if (iter == 0) {
    int va = visited_time[(size_t)b * GS + s_action];
    for (int g = t; g < GS; g += 256) {
      int d = visited_time[(size_t)b * GS + g] - va;
      vt[g] = ((d % GS) + GS) % GS;
    }
  }
  __syncthreads();
  if (t == 0) s_vtt_a = vt[s_action];
  __syncthreads();

  // 5) mask update
  const int allow = (!s_stop_new) && (s_nna == s_ai0);
  for (int g = t; g < GS; g += 256) {
    int mnew = (vt[g] <= s_vtt_a);
    if (iter == 0) mnew |= (vt[g] > GS - 2);
    if (s_stop_new && g == s_action) mnew = 0;
    if (allow && g == s_ai0) mnew = 0;
    mk[g] = mnew;
  }

  // 6) next inputs
  if (t < EE) {
    float x1 = h[((size_t)b * GS + s_action) * EE + t];
    iq1[b * EE + t] = x1;
    int n2 = ((s_nol_old % GS) + GS) % GS;
    iq2[b * EE + t] = s_stop_old ? x1 : h[((size_t)b * GS + n2) * EE + t];
  }
}

// ---------------------------------------------------------------- finalize outputs
__global__ __launch_bounds__(BSZ) void k_final(const int* __restrict__ stopped,
                                               const int* __restrict__ ai,
                                               const int* __restrict__ kl,
                                               const int* __restrict__ kr,
                                               const float* __restrict__ ll,
                                               float* __restrict__ out) {
  int b = threadIdx.x;
  if (b >= BSZ) return;
  int kr3 = kr[b * KMAX + 3];
  if (!stopped[b]) kr3 = kl[b * (KMAX + 1) + KMAX];
  float* o = out + b * 12;
  for (int j = 0; j < 4; ++j) o[j] = (float)ai[b * KMAX + j];
  for (int j = 0; j < 4; ++j) o[4 + j] = (float)kl[b * (KMAX + 1) + j];
  o[8] = (float)kr[b * KMAX + 0];
  o[9] = (float)kr[b * KMAX + 1];
  o[10] = (float)kr[b * KMAX + 2];
  o[11] = (float)kr3;
  out[BSZ * 12 + b] = ll[b];
}

// ================================================================ launch
extern "C" void kernel_launch(void* const* d_in, const int* in_sizes, int n_in,
                              void* d_out, int out_size, void* d_ws, size_t ws_size,
                              hipStream_t stream) {
  const float* h             = (const float*)d_in[0];
  const int*   rec           = (const int*)d_in[1];
  const int*   visited_time  = (const int*)d_in[3];
  const int*   last_action   = (const int*)d_in[4];
  const int*   fixed_action  = (const int*)d_in[5];
  const float* W_K1 = (const float*)d_in[6];
  const float* W_K2 = (const float*)d_in[7];
  const float* W_K3 = (const float*)d_in[8];
  const float* W_K4 = (const float*)d_in[9];
  const float* W_Q1 = (const float*)d_in[10];
  const float* W_Q2 = (const float*)d_in[11];
  const float* W_Q3 = (const float*)d_in[12];
  const float* W_Q4 = (const float*)d_in[13];
  const float* W_init     = (const float*)d_in[14];
  const float* b_init     = (const float*)d_in[15];
  const float* V1         = (const float*)d_in[16];
  const float* V2         = (const float*)d_in[17];
  const float* init_query = (const float*)d_in[18];
  const float* r1_Wih = (const float*)d_in[19];
  const float* r1_Whh = (const float*)d_in[20];
  const float* r1_bih = (const float*)d_in[21];
  const float* r1_bhh = (const float*)d_in[22];
  const float* r2_Wih = (const float*)d_in[23];
  const float* r2_Whh = (const float*)d_in[24];
  const float* r2_bih = (const float*)d_in[25];
  const float* r2_bhh = (const float*)d_in[26];

  char* ws = (char*)d_ws;
  size_t off = 0;
  auto alloc = [&](size_t bytes) { void* p = ws + off; off += (bytes + 255) & ~(size_t)255; return p; };
  unsigned short* wKbf = (unsigned short*)alloc(4 * EE * EE * sizeof(unsigned short));
  float* hmean = (float*)alloc(BSZ * EE * sizeof(float));
  float* q1    = (float*)alloc(BSZ * EE * sizeof(float));
  float* q2    = (float*)alloc(BSZ * EE * sizeof(float));
  float* iq1   = (float*)alloc(BSZ * EE * sizeof(float));
  float* iq2   = (float*)alloc(BSZ * EE * sizeof(float));
  float* A1q   = (float*)alloc(BSZ * EE * sizeof(float));
  float* B1q   = (float*)alloc(BSZ * EE * sizeof(float));
  float* A2q   = (float*)alloc(BSZ * EE * sizeof(float));
  float* B2q   = (float*)alloc(BSZ * EE * sizeof(float));
  float* logits = (float*)alloc((size_t)BSZ * GS * sizeof(float));
  int* vtt     = (int*)alloc((size_t)BSZ * GS * sizeof(int));
  int* mask    = (int*)alloc((size_t)BSZ * GS * sizeof(int));
  int* stopped = (int*)alloc(BSZ * sizeof(int));
  int* nol     = (int*)alloc(BSZ * sizeof(int));
  int* ai      = (int*)alloc(BSZ * KMAX * sizeof(int));
  int* kl      = (int*)alloc(BSZ * (KMAX + 1) * sizeof(int));
  int* kr      = (int*)alloc(BSZ * KMAX * sizeof(int));
  float* ll    = (float*)alloc(BSZ * sizeof(float));

  k_prep<<<(4 * EE * EE + 255) / 256, 256, 0, stream>>>(W_K1, W_K2, W_K3, W_K4, wKbf);
  k_init<<<(BSZ * GS + 255) / 256, 256, 0, stream>>>(mask, stopped, nol, ll, ai, kl, kr);
  k_mean<<<BSZ, EE, 0, stream>>>(h, hmean);
  k_qinit<<<BSZ, EE, 0, stream>>>(hmean, W_init, b_init, init_query, q1, q2, iq1, iq2);

  const size_t wlds_bytes = (size_t)WLDS_DW * sizeof(int);   // 139264 B dynamic LDS
  for (int i = 0; i < KMAX; ++i) {
    k_gru<<<BSZ, EE, 0, stream>>>(r1_Wih, r1_Whh, r1_bih, r1_bhh,
                                  r2_Wih, r2_Whh, r2_bih, r2_bhh,
                                  W_Q1, W_Q2, W_Q3, W_Q4,
                                  iq1, iq2, q1, q2, A1q, B1q, A2q, B2q);
    k_score<<<dim3(BSZ, 16), 256, wlds_bytes, stream>>>(h, wKbf, A1q, B1q, A2q, B2q, V1, V2, logits);
    k_select<<<BSZ, 256, 0, stream>>>(i, h, rec, visited_time, last_action, fixed_action,
                                      logits, mask, vtt, stopped, nol, ai, kl, kr, ll, iq1, iq2);
  }
  k_final<<<1, BSZ, 0, stream>>>(stopped, ai, kl, kr, ll, (float*)d_out);
}
